// TransformerBlockBackbone_35802847379570
// MI455X (gfx1250) — compile-verified, matrix operand order checked by reading, not run on hardware
//
#include <hip/hip_runtime.h>
#include <hip/hip_bf16.h>

typedef __attribute__((ext_vector_type(16))) _Float16 v16h;
typedef __attribute__((ext_vector_type(8)))  _Float16 v8h;
typedef __attribute__((ext_vector_type(8)))  float    v8f;
typedef int v4i __attribute__((vector_size(16)));   // matches builtin param type

#define NPTS   (2 * 16384)   // B*N
#define NPER   16384         // N
#define DM     128           // d_model
#define DP     64            // d_points
#define KNB    16            // neighbors
#define WPB2   8             // waves per block, kernel 2

#if defined(__has_builtin)
#if __has_builtin(__builtin_amdgcn_global_load_async_to_lds_b128) && \
    __has_builtin(__builtin_amdgcn_s_wait_asynccnt)
#define USE_ASYNC_LDS 1
#endif
#endif
#ifndef USE_ASYNC_LDS
#define USE_ASYNC_LDS 0
#endif

// ---------------------------------------------------------------------------
// WMMA helper: D = A(16x32 f16) * B(32x16 f16) + C(16x16 f32)
// ---------------------------------------------------------------------------
__device__ __forceinline__ v8f wmma_f16(v16h a, v16h b, v8f c) {
  return __builtin_amdgcn_wmma_f32_16x16x32_f16(
      /*neg_a=*/false, a, /*neg_b=*/false, b,
      /*c_mod=*/(short)0, c, /*reuse_a=*/false, /*reuse_b=*/false);
}

// 16-byte global -> LDS copy: async DMA (ASYNCcnt) when available.
__device__ __forceinline__ void cp_b128(const _Float16* g, _Float16* l) {
#if USE_ASYNC_LDS
  __builtin_amdgcn_global_load_async_to_lds_b128(
      (__attribute__((address_space(1))) v4i*)g,
      (__attribute__((address_space(3))) v4i*)l, 0, 0);
#else
  *(v8h*)l = *(const v8h*)g;
#endif
}

__device__ __forceinline__ void cp_wait() {
#if USE_ASYNC_LDS
  __builtin_amdgcn_s_wait_asynccnt(0);
#endif
  asm volatile("s_wait_dscnt 0x0" ::: "memory");
}

// Fragment load from f16 memory (LDS or global), row base pointer, K-offset.
// Per-lane layout (A and B symmetric): kb = (lane&16)?8:0
//   halves 0..7  -> k = koff + kb + (0..7)
//   halves 8..15 -> k = koff + 16 + kb + (0..7)
__device__ __forceinline__ v16h frag_ld(const _Float16* row, int koff, int kb) {
  v8h lo = *(const v8h*)(row + koff + kb);
  v8h hi = *(const v8h*)(row + koff + kb + 16);
  return __builtin_shufflevector(lo, hi, 0, 1, 2, 3, 4, 5, 6, 7,
                                 8, 9, 10, 11, 12, 13, 14, 15);
}

// Fragment sourced from f32 global memory with on-the-fly cvt (inputs only).
__device__ __forceinline__ v16h frag_ld_f32(const float* row, int koff, int kb) {
  const float* p = row + koff + kb;
  const float* q = row + koff + kb + 16;
  v16h r;
#pragma unroll
  for (int i = 0; i < 8; ++i) {
    r[i]     = (_Float16)p[i];
    r[i + 8] = (_Float16)q[i];
  }
  return r;
}

// ---------------------------------------------------------------------------
// Kernel 0: one-time f32 -> f16 weight conversion into workspace
// ---------------------------------------------------------------------------
__global__ void cvt_kernel(const float* __restrict__ s, _Float16* __restrict__ d,
                           int n) {
  int i = blockIdx.x * blockDim.x + threadIdx.x;
  if (i < n) d[i] = (_Float16)s[i];
}

// ---------------------------------------------------------------------------
// Kernel 1: x = fc1(features);  q,k,v = x @ W{q,k,v}^T   (stored f16 in ws)
// One wave per 16-row tile of [B*N, .]; weights read as pre-converted f16.
// ---------------------------------------------------------------------------
__device__ __forceinline__ void proj_store(const v16h xf[4],
                                           const _Float16* __restrict__ W,
                                           _Float16* __restrict__ O, int row0,
                                           int lane) {
  const int ml  = lane & 15;
  const int kb  = (lane & 16) >> 1;
  const int hi8 = (lane >> 4) << 3;
  for (int t = 0; t < 8; ++t) {
    v8f acc = {};
    const _Float16* wr = W + (t * 16 + ml) * DM;
#pragma unroll
    for (int fi = 0; fi < 4; ++fi)
      acc = wmma_f16(xf[fi], frag_ld(wr, fi * 32, kb), acc);
    const int f = t * 16 + ml;
#pragma unroll
    for (int r = 0; r < 8; ++r)
      O[(row0 + r + hi8) * DM + f] = (_Float16)acc[r];
  }
}

__global__ void __launch_bounds__(128)
proj_kernel(const float* __restrict__ feat, const _Float16* __restrict__ fc1w16,
            const float* __restrict__ fc1b, const _Float16* __restrict__ wq16,
            const _Float16* __restrict__ wk16, const _Float16* __restrict__ wv16,
            _Float16* __restrict__ qo, _Float16* __restrict__ ko,
            _Float16* __restrict__ vo) {
  __shared__ __align__(16) _Float16 xstage[4 * 16 * DM];  // 16 KB, per-wave tiles

  const int lane = threadIdx.x & 31;
  const int wid  = threadIdx.x >> 5;
  const int row0 = (blockIdx.x * 4 + wid) * 16;
  const int ml   = lane & 15;
  const int kb   = (lane & 16) >> 1;
  const int hi8  = (lane >> 4) << 3;
  _Float16* xs = xstage + wid * 16 * DM;

  // A fragments of features tile (16 x 64), f32 -> f16 once
  v16h af[2];
  af[0] = frag_ld_f32(feat + (row0 + ml) * DP, 0, kb);
  af[1] = frag_ld_f32(feat + (row0 + ml) * DP, 32, kb);

  // x = feat @ fc1w^T + fc1b  -> LDS (f16, row-major [16][128])
  for (int t = 0; t < 8; ++t) {
    v8f acc = {};
    const _Float16* wr = fc1w16 + (t * 16 + ml) * DP;
    acc = wmma_f16(af[0], frag_ld(wr, 0, kb), acc);
    acc = wmma_f16(af[1], frag_ld(wr, 32, kb), acc);
    const int f    = t * 16 + ml;
    const float bb = fc1b[f];
#pragma unroll
    for (int r = 0; r < 8; ++r)
      xs[(r + hi8) * DM + f] = (_Float16)(acc[r] + bb);
  }
  asm volatile("s_wait_dscnt 0x0" ::: "memory");  // wave-private LDS RAW

  // Re-read x in A-fragment layout (16 x 128 -> 4 fragments)
  v16h xf[4];
#pragma unroll
  for (int fi = 0; fi < 4; ++fi) xf[fi] = frag_ld(xs + ml * DM, fi * 32, kb);

  proj_store(xf, wq16, qo, row0, lane);
  proj_store(xf, wk16, ko, row0, lane);
  proj_store(xf, wv16, vo, row0, lane);
}

// ---------------------------------------------------------------------------
// Kernel 2: fused per-point transformer attention. One wave per point.
// LDS: [dw2|gw1|gw2] f16 (96 KB, async-preloaded) + 8 waves * 8 KB scratch.
// Gathered k/v neighbor rows are DMA-staged into the wave scratch and updated
// in place to h = q-k+pos and vpe = v+pos.
// ---------------------------------------------------------------------------
__global__ void __launch_bounds__(WPB2 * 32)
attn_kernel(const float* __restrict__ new_xyz, const float* __restrict__ gxyz,
            const int* __restrict__ gidx, const float* __restrict__ dw1,
            const float* __restrict__ db1, const float* __restrict__ db2,
            const float* __restrict__ gb1, const float* __restrict__ gb2,
            const _Float16* __restrict__ w3,  // [dw2|gw1|gw2] f16, contiguous
            const _Float16* __restrict__ qi, const _Float16* __restrict__ ki,
            const _Float16* __restrict__ vi, float* __restrict__ out) {
  extern __shared__ _Float16 smem[];
  _Float16* WdW2 = smem;                 // 128*128 f16
  _Float16* WgW1 = smem + DM * DM;
  _Float16* WgW2 = smem + 2 * DM * DM;
  _Float16* scr  = smem + 3 * DM * DM;   // WPB2 waves * 2 * 16*128 f16

  // Cooperative weight preload: f16 global -> LDS via async DMA
  for (int i = threadIdx.x; i < 3 * DM * DM / 8; i += blockDim.x)
    cp_b128(w3 + i * 8, smem + i * 8);
  cp_wait();
  __syncthreads();

  const int lane = threadIdx.x & 31;
  const int wid  = threadIdx.x >> 5;
  const int pt   = blockIdx.x * WPB2 + wid;
  const int b    = pt / NPER;
  const int n    = pt % NPER;
  const int ml   = lane & 15;         // neighbor row (A) / feature col (B,C)
  const int kb   = (lane & 16) >> 1;  // 0 or 8
  const int hi8  = (lane >> 4) << 3;  // C rows 0-7 vs 8-15

  _Float16* hbuf = scr + wid * (2 * 16 * DM);  // staged k rows -> h
  _Float16* vbuf = hbuf + 16 * DM;             // staged v rows -> vpe

  // Stage the 16 gathered k/v rows (256 B each) into LDS via async DMA.
  // Per pass: 2 rows; lane -> (row = lane>>4, 16B chunk = lane&15).
  {
    const int rsel  = lane >> 4;
    const int chunk = (lane & 15) * 8;
#pragma unroll
    for (int p = 0; p < 8; ++p) {
      const int m   = p * 2 + rsel;
      const int idx = gidx[pt * KNB + m];
      const int src = (b * NPER + idx) * DM + chunk;
      cp_b128(ki + src, hbuf + m * DM + chunk);
      cp_b128(vi + src, vbuf + m * DM + chunk);
    }
  }

  // Overlap with DMA: rel + pos1 = relu(rel @ dw1^T + db1) in A-frag layout
  const float* nx = new_xyz + (b * NPER + n) * 3;
  const float r0 = nx[0] - gxyz[((b * 3 + 0) * NPER + n) * KNB + ml];
  const float r1 = nx[1] - gxyz[((b * 3 + 1) * NPER + n) * KNB + ml];
  const float r2 = nx[2] - gxyz[((b * 3 + 2) * NPER + n) * KNB + ml];

  v16h pf[4];
#pragma unroll
  for (int fi = 0; fi < 4; ++fi) {
#pragma unroll
    for (int j = 0; j < 16; ++j) {
      const int kl = (j < 8) ? (kb + j) : (16 + kb + (j - 8));
      const int f  = fi * 32 + kl;
      float v = db1[f] + r0 * dw1[f * 3] + r1 * dw1[f * 3 + 1] + r2 * dw1[f * 3 + 2];
      pf[fi][j] = (_Float16)fmaxf(v, 0.f);
    }
  }

  cp_wait();  // k/v rows staged

  const _Float16* qrow = qi + pt * DM;

  // pos_enc = pos1 @ dw2^T + db2; in-place: h = q-k+pos, vpe = v+pos
  for (int t = 0; t < 8; ++t) {
    v8f acc = {};
    const _Float16* wr = WdW2 + (t * 16 + ml) * DM;
#pragma unroll
    for (int fi = 0; fi < 4; ++fi)
      acc = wmma_f16(pf[fi], frag_ld(wr, fi * 32, kb), acc);
    const int f    = t * 16 + ml;
    const float bb = db2[f];
    const float qf = (float)qrow[f];
#pragma unroll
    for (int r = 0; r < 8; ++r) {
      const float pos = acc[r] + bb;
      const int o     = (r + hi8) * DM + f;
      const float kf  = (float)hbuf[o];
      const float vf  = (float)vbuf[o];
      hbuf[o] = (_Float16)(qf - kf + pos);
      vbuf[o] = (_Float16)(vf + pos);
    }
  }
  asm volatile("s_wait_dscnt 0x0" ::: "memory");

  v16h hf[4];
#pragma unroll
  for (int fi = 0; fi < 4; ++fi) hf[fi] = frag_ld(hbuf + ml * DM, fi * 32, kb);

  // g = relu(h @ gw1^T + gb1) -> overwrite hbuf
  for (int t = 0; t < 8; ++t) {
    v8f acc = {};
    const _Float16* wr = WgW1 + (t * 16 + ml) * DM;
#pragma unroll
    for (int fi = 0; fi < 4; ++fi)
      acc = wmma_f16(hf[fi], frag_ld(wr, fi * 32, kb), acc);
    const int f    = t * 16 + ml;
    const float bb = gb1[f];
#pragma unroll
    for (int r = 0; r < 8; ++r)
      hbuf[(r + hi8) * DM + f] = (_Float16)fmaxf(acc[r] + bb, 0.f);
  }
  asm volatile("s_wait_dscnt 0x0" ::: "memory");

  v16h gf[4];
#pragma unroll
  for (int fi = 0; fi < 4; ++fi) gf[fi] = frag_ld(hbuf + ml * DM, fi * 32, kb);

  // attn = g @ gw2^T + gb2; softmax over 16 neighbors; out = sum attn*vpe
  const float inv = 0.08838834764831845f;  // 1/sqrt(128)
  for (int t = 0; t < 8; ++t) {
    v8f acc = {};
    const _Float16* wr = WgW2 + (t * 16 + ml) * DM;
#pragma unroll
    for (int fi = 0; fi < 4; ++fi)
      acc = wmma_f16(gf[fi], frag_ld(wr, fi * 32, kb), acc);
    const int f    = t * 16 + ml;
    const float bb = gb2[f];
    float z[8];
    float mx = -3.0e38f;
#pragma unroll
    for (int r = 0; r < 8; ++r) {
      z[r] = (acc[r] + bb) * inv;
      mx   = fmaxf(mx, z[r]);
    }
    mx = fmaxf(mx, __shfl_xor(mx, 16, 32));  // combine rows 0-7 with 8-15
    float s = 0.f;
#pragma unroll
    for (int r = 0; r < 8; ++r) {
      z[r] = __expf(z[r] - mx);
      s += z[r];
    }
    s += __shfl_xor(s, 16, 32);
    const float rs = 1.f / s;
    float part = 0.f;
#pragma unroll
    for (int r = 0; r < 8; ++r)
      part += z[r] * rs * (float)vbuf[(r + hi8) * DM + f];
    part += __shfl_xor(part, 16, 32);
    if (lane < 16) out[pt * DM + f] = part;
  }
}

// ---------------------------------------------------------------------------
extern "C" void kernel_launch(void* const* d_in, const int* in_sizes, int n_in,
                              void* d_out, int out_size, void* d_ws, size_t ws_size,
                              hipStream_t stream) {
  (void)in_sizes; (void)n_in; (void)out_size; (void)ws_size;

  const float* new_xyz = (const float*)d_in[0];
  const float* gxyz    = (const float*)d_in[1];
  const int*   gidx    = (const int*)d_in[2];
  const float* feat    = (const float*)d_in[3];
  const float* fc1w    = (const float*)d_in[4];
  const float* fc1b    = (const float*)d_in[5];
  const float* wq      = (const float*)d_in[6];
  const float* wk      = (const float*)d_in[7];
  const float* wv      = (const float*)d_in[8];
  const float* dw1     = (const float*)d_in[9];
  const float* db1     = (const float*)d_in[10];
  const float* dw2     = (const float*)d_in[11];
  const float* db2     = (const float*)d_in[12];
  const float* gw1     = (const float*)d_in[13];
  const float* gb1     = (const float*)d_in[14];
  const float* gw2     = (const float*)d_in[15];
  const float* gb2     = (const float*)d_in[16];
  float* out = (float*)d_out;

  // Workspace layout (f16): q/k/v projections + pre-converted weights
  _Float16* qws    = (_Float16*)d_ws;
  _Float16* kws    = qws + (size_t)NPTS * DM;
  _Float16* vws    = kws + (size_t)NPTS * DM;
  _Float16* fc1w16 = vws + (size_t)NPTS * DM;
  _Float16* wq16   = fc1w16 + DM * DP;
  _Float16* wk16   = wq16 + DM * DM;
  _Float16* wv16   = wk16 + DM * DM;
  _Float16* w3     = wv16 + DM * DM;  // [dw2|gw1|gw2] contiguous

  // Kernel 0: one-time weight conversion (f32 -> f16)
  cvt_kernel<<<(DM * DP + 255) / 256, 256, 0, stream>>>(fc1w, fc1w16, DM * DP);
  cvt_kernel<<<(DM * DM + 255) / 256, 256, 0, stream>>>(wq, wq16, DM * DM);
  cvt_kernel<<<(DM * DM + 255) / 256, 256, 0, stream>>>(wk, wk16, DM * DM);
  cvt_kernel<<<(DM * DM + 255) / 256, 256, 0, stream>>>(wv, wv16, DM * DM);
  cvt_kernel<<<(DM * DM + 255) / 256, 256, 0, stream>>>(dw2, w3, DM * DM);
  cvt_kernel<<<(DM * DM + 255) / 256, 256, 0, stream>>>(gw1, w3 + DM * DM, DM * DM);
  cvt_kernel<<<(DM * DM + 255) / 256, 256, 0, stream>>>(gw2, w3 + 2 * DM * DM, DM * DM);

  // Kernel 1: 2048 row-tiles, 4 waves/block
  proj_kernel<<<NPTS / 16 / 4, 128, 0, stream>>>(feat, fc1w16, fc1b, wq16, wk16,
                                                 wv16, qws, kws, vws);

  // Kernel 2: one wave per point, 8 points/block; 160 KB dynamic LDS
  const size_t shmem =
      (size_t)(3 * DM * DM + WPB2 * 2 * 16 * DM) * sizeof(_Float16);
  (void)hipFuncSetAttribute((const void*)&attn_kernel,
                            hipFuncAttributeMaxDynamicSharedMemorySize,
                            (int)shmem);
  attn_kernel<<<NPTS / WPB2, WPB2 * 32, shmem, stream>>>(
      new_xyz, gxyz, gidx, dw1, db1, db2, gb1, gb2, w3, qws, kws, vws, out);
}
